// StackedSTU1D_48550310314368
// MI455X (gfx1250) — compile-verified, matrix operand order checked by reading, not run on hardware
//
#include <hip/hip_runtime.h>
#include <hip/hip_bf16.h>
#include <math.h>

// StackedSTU1D on gfx1250 (MI455X), wave32 + WMMA bf16.
// Haar top-K structure exploited: spectral block == 15-tap circular conv on dz
// plus a global-mean rank-1 term.  All GEMMs via v_wmma_f32_16x16x32_bf16.
//
// Shapes: B=16, L=1024, CIN=3, D=128, K=16, DEPTH=4, COUT=1.

typedef __bf16 bf16_t;
typedef __attribute__((ext_vector_type(16))) __bf16 v16bf;
typedef __attribute__((ext_vector_type(8)))  __bf16 v8bf;
typedef __attribute__((ext_vector_type(4)))  __bf16 v4bf;
typedef __attribute__((ext_vector_type(8)))  float  v8f;
typedef __attribute__((ext_vector_type(4)))  float  v4f;

#define WMMA_BF16(A_, B_, C_) \
  __builtin_amdgcn_wmma_f32_16x16x32_bf16(false, (A_), false, (B_), (short)0, (C_), false, false)

#define B_   16
#define L_   1024
#define D_   128
#define DEPTH_ 4
#define INV_SQRT2 0.70710678118654752f

// ---- workspace layout (bytes, 256-aligned) -------------------------------
#define OFF_H     0u                       // f32  h     (B,L,D)   8 MiB
#define OFF_Z     8388608u                 // bf16 z     (B,L,D)   4 MiB
#define OFF_PART  12582912u                // f32  part  (B,16,D)  128 KiB
#define OFF_S0    12713984u                // f32  S0    (B,D)     8 KiB
#define OFF_CF    12722176u                // bf16 conv frags 4*245760
#define OFF_F1    14688256u                // bf16 fc1  frags 4*32768
#define OFF_F2    14950400u                // bf16 fc2  frags 4*32768
// total ~15.2 MiB

#define CF_PER_LAYER 245760   // [15 taps][4 dchunk][8 hblk][32 lane][16]
#define F1_PER_LAYER 32768    // [4 dchunk][16 h1blk][32 lane][16]
#define F2_PER_LAYER 32768    // [8 dchunk][8 hblk][32 lane][16]
#define PREP_PER_LAYER (CF_PER_LAYER + F1_PER_LAYER + F2_PER_LAYER) // 311296

// ---------------------------------------------------------------------------
// Pack theta / fc1_w / fc2_w into WMMA B-fragment layout (bf16).
// B-matrix (32x16, 16-bit): lane = {half = lane>>4, ncol = lane&15};
//   per-lane 16 values j: K = (j<8) ? half*8+j : 16 + half*8 + (j-8).
// ---------------------------------------------------------------------------
__global__ void stu_prep_frags(const float* __restrict__ theta,
                               const float* __restrict__ fc1w,
                               const float* __restrict__ fc2w,
                               bf16_t* __restrict__ cf,
                               bf16_t* __restrict__ f1,
                               bf16_t* __restrict__ f2)
{
  int idx = blockIdx.x * 256 + threadIdx.x;
  if (idx >= DEPTH_ * PREP_PER_LAYER) return;
  int layer = idx / PREP_PER_LAYER;
  int r     = idx % PREP_PER_LAYER;

  if (r < CF_PER_LAYER) {
    int t  = r / 16384;       int r2 = r % 16384;   // tap 0..14 -> haar k = t+1
    int c  = r2 / 4096;       int r3 = r2 % 4096;   // d-chunk
    int w  = r3 / 512;        int r4 = r3 % 512;    // h block
    int lane = r4 >> 4;       int j  = r4 & 15;
    int half = lane >> 4;     int ncol = lane & 15;
    int kk = (j < 8) ? (half * 8 + j) : (16 + half * 8 + (j - 8));
    int h  = w * 16 + ncol;
    int d  = c * 32 + kk;
    float v = INV_SQRT2 *
        theta[(((size_t)(layer * 16 + (t + 1)) * D_ + h)) * D_ + d];
    cf[(size_t)layer * CF_PER_LAYER + r] = (bf16_t)v;
  } else if (r < CF_PER_LAYER + F1_PER_LAYER) {
    int q  = r - CF_PER_LAYER;
    int c  = q / 8192;        int r3 = q % 8192;
    int blk = r3 / 512;       int r4 = r3 % 512;
    int lane = r4 >> 4;       int j  = r4 & 15;
    int half = lane >> 4;     int ncol = lane & 15;
    int kk = (j < 8) ? (half * 8 + j) : (16 + half * 8 + (j - 8));
    int h1 = blk * 16 + ncol;
    int d  = c * 32 + kk;
    float v = fc1w[((size_t)(layer * 256 + h1)) * D_ + d];
    f1[(size_t)layer * F1_PER_LAYER + q] = (bf16_t)v;
  } else {
    int q  = r - CF_PER_LAYER - F1_PER_LAYER;
    int c  = q / 4096;        int r3 = q % 4096;
    int blk = r3 / 512;       int r4 = r3 % 512;
    int lane = r4 >> 4;       int j  = r4 & 15;
    int half = lane >> 4;     int ncol = lane & 15;
    int kk = (j < 8) ? (half * 8 + j) : (16 + half * 8 + (j - 8));
    int h  = blk * 16 + ncol;
    int dd = c * 32 + kk;
    float v = fc2w[((size_t)(layer * D_ + h)) * 256 + dd];
    f2[(size_t)layer * F2_PER_LAYER + q] = (bf16_t)v;
  }
}

// ---------------------------------------------------------------------------
// Lift: h[b,l,d] = sum_c lift_w[d,c] * xg[b,c,l] + lift_b[d]; coord channel.
// ---------------------------------------------------------------------------
__global__ void stu_lift(const float* __restrict__ x,
                         const float* __restrict__ lw,
                         const float* __restrict__ lb,
                         float* __restrict__ h)
{
  int t = blockIdx.x * 256 + threadIdx.x;
  if (t >= B_ * L_) return;
  int b = t >> 10, l = t & (L_ - 1);
  float c0 = x[((size_t)(b * 3 + 0)) * L_ + l];
  float c1 = x[((size_t)(b * 3 + 1)) * L_ + l];
  float c2 = x[((size_t)(b * 3 + 2)) * L_ + l];
  float c3 = (float)l * (1.0f / (float)(L_ - 1));
  float* hp = h + (size_t)t * D_;
#pragma unroll 4
  for (int d = 0; d < D_; ++d) {
    const float* wd = lw + d * 4;
    hp[d] = wd[0] * c0 + wd[1] * c1 + wd[2] * c2 + wd[3] * c3 + lb[d];
  }
}

// ---------------------------------------------------------------------------
// LayerNorm over channels (D) per (b,l) column; writes z bf16 (B,L,D) and
// per-block partial channel sums for the Haar k=0 (mean) term.
// block = 256 threads (8 waves); block handles 64 consecutive l.
// ---------------------------------------------------------------------------
__global__ void stu_ln(const float* __restrict__ h,
                       const float* __restrict__ ln_g,
                       const float* __restrict__ ln_b,
                       bf16_t* __restrict__ z,
                       float* __restrict__ part, int layer)
{
  int b  = blockIdx.x >> 4;
  int blk = blockIdx.x & 15;
  int l0 = blk * 64;
  int w    = threadIdx.x >> 5;
  int lane = threadIdx.x & 31;
  __shared__ float red[8 * D_];

  v4f g  = *(const v4f*)(ln_g + layer * D_ + lane * 4);
  v4f be = *(const v4f*)(ln_b + layer * D_ + lane * 4);
  v4f acc = {0.f, 0.f, 0.f, 0.f};

  for (int q = 0; q < 8; ++q) {
    int l = l0 + w * 8 + q;
    const float* col = h + ((size_t)(b * L_ + l)) * D_;
    v4f v = *(const v4f*)(col + lane * 4);
    float s  = v.x + v.y + v.z + v.w;
    float ss = v.x * v.x + v.y * v.y + v.z * v.z + v.w * v.w;
#pragma unroll
    for (int m = 1; m < 32; m <<= 1) {
      s  += __shfl_xor(s, m, 32);
      ss += __shfl_xor(ss, m, 32);
    }
    float mu  = s * (1.0f / 128.0f);
    float var = ss * (1.0f / 128.0f) - mu * mu;
    float rs  = rsqrtf(var + 1e-5f);
    v4f zz;
    zz.x = (v.x - mu) * rs * g.x + be.x;
    zz.y = (v.y - mu) * rs * g.y + be.y;
    zz.z = (v.z - mu) * rs * g.z + be.z;
    zz.w = (v.w - mu) * rs * g.w + be.w;
    v4bf zb = { (bf16_t)zz.x, (bf16_t)zz.y, (bf16_t)zz.z, (bf16_t)zz.w };
    *(v4bf*)(z + ((size_t)(b * L_ + l)) * D_ + lane * 4) = zb;
    acc.x += zz.x; acc.y += zz.y; acc.z += zz.z; acc.w += zz.w;
  }
  *(v4f*)(red + w * D_ + lane * 4) = acc;
  __syncthreads();
  if (threadIdx.x < D_) {
    float s = 0.f;
#pragma unroll
    for (int w2 = 0; w2 < 8; ++w2) s += red[w2 * D_ + threadIdx.x];
    part[((size_t)(b * 16 + blk)) * D_ + threadIdx.x] = s;
  }
}

// S0[b,h] = (1/sqrt(L)) * sum_d theta[layer,0,h,d] * sum_l z[b,d,l]
__global__ void stu_s0(const float* __restrict__ part,
                       const float* __restrict__ theta,
                       float* __restrict__ S0, int layer)
{
  int b = blockIdx.x, t = threadIdx.x;   // 128 threads
  __shared__ float zs[D_];
  float s = 0.f;
#pragma unroll
  for (int k = 0; k < 16; ++k) s += part[((size_t)(b * 16 + k)) * D_ + t];
  zs[t] = s;
  __syncthreads();
  const float* th0 = theta + (size_t)layer * 16 * D_ * D_;  // k = 0 slice
  float a = 0.f;
  for (int d = 0; d < D_; ++d) a += th0[t * D_ + d] * zs[d];
  S0[b * D_ + t] = a * (1.0f / 32.0f);   // 1/sqrt(1024)
}

// ---------------------------------------------------------------------------
// Fused spectral-conv + MLP + residual for one (b, 64-wide l tile).
// 8 waves, each owns one 16-wide h block across 4 l-tiles of 16.
// LDS: dz tile [92][136] bf16 (overlaid later by H1 [64][264]); S [64][136].
// ---------------------------------------------------------------------------
#define ZSTR 136
#define SSTR 136
#define HSTR 264

__global__ __launch_bounds__(256, 1)
void stu_layer(float* __restrict__ h,
               const bf16_t* __restrict__ z,
               const float* __restrict__ S0,
               const bf16_t* __restrict__ cf,
               const bf16_t* __restrict__ f1,
               const bf16_t* __restrict__ f2,
               const float* __restrict__ fc1b,
               const float* __restrict__ fc2b, int layer)
{
  __shared__ __align__(16) bf16_t region0[64 * HSTR];  // dz tile, later H1
  __shared__ __align__(16) bf16_t bufA[64 * SSTR];     // S (conv output)
  bf16_t* zt   = region0;   // [92][ZSTR] (92*136 = 12512 <= 64*264)
  bf16_t* bufB = region0;   // [64][HSTR] after conv phase

  int b  = blockIdx.x >> 4;
  int l0 = (blockIdx.x & 15) * 64;
  const bf16_t* zb = z + (size_t)b * L_ * D_;

  // ---- stage dz = z[l] - z[l+1] (circular), rows 0..91 ----
  for (int it = threadIdx.x; it < 92 * 32; it += 256) {
    int row = it >> 5, ch = it & 31;
    int lA = (l0 + row) & (L_ - 1);
    int lB = (l0 + row + 1) & (L_ - 1);
    v4bf a0 = *(const v4bf*)(zb + (size_t)lA * D_ + ch * 4);
    v4bf a1 = *(const v4bf*)(zb + (size_t)lB * D_ + ch * 4);
    v4bf dz;
#pragma unroll
    for (int i = 0; i < 4; ++i) dz[i] = (bf16_t)((float)a0[i] - (float)a1[i]);
    *(v4bf*)(zt + row * ZSTR + ch * 4) = dz;
  }
  __syncthreads();

  int w    = threadIdx.x >> 5;
  int lane = threadIdx.x & 31;
  int half = lane >> 4;
  int ncol = lane & 15;
  int m    = ncol;             // A-fragment row

  // ---- conv: S = S0 + sum_taps sum_d theta_frag * dz ----
  float s0v = S0[b * D_ + w * 16 + ncol];
  v8f acc[4];
#pragma unroll
  for (int j = 0; j < 4; ++j)
#pragma unroll
    for (int e = 0; e < 8; ++e) acc[j][e] = s0v;

  const bf16_t* cfl = cf + (size_t)layer * CF_PER_LAYER;
  for (int t = 0; t < 15; ++t) {
    int o = 2 * t;
#pragma unroll
    for (int c = 0; c < 4; ++c) {
      v16bf bfr = *(const v16bf*)(cfl + (size_t)(((t * 4 + c) * 8 + w) * 512 + lane * 16));
      int dbase = c * 32 + half * 8;
#pragma unroll
      for (int j = 0; j < 4; ++j) {
        int x = j * 16 + m + o;
        v8bf lo = *(const v8bf*)(zt + x * ZSTR + dbase);
        v8bf hi = *(const v8bf*)(zt + x * ZSTR + dbase + 16);
        v16bf af = __builtin_shufflevector(lo, hi, 0,1,2,3,4,5,6,7,8,9,10,11,12,13,14,15);
        acc[j] = WMMA_BF16(af, bfr, acc[j]);
      }
    }
  }
  // write S (bf16) to bufA
#pragma unroll
  for (int j = 0; j < 4; ++j)
#pragma unroll
    for (int r = 0; r < 8; ++r)
      bufA[(j * 16 + r + half * 8) * SSTR + (w * 16 + ncol)] = (bf16_t)acc[j][r];
  __syncthreads();

  // ---- fc1 + GELU: H1 (64 x 256) ----
  v8f acc2[2][4];
#pragma unroll
  for (int p = 0; p < 2; ++p) {
    float bv = fc1b[layer * 256 + (w * 2 + p) * 16 + ncol];
#pragma unroll
    for (int j = 0; j < 4; ++j)
#pragma unroll
      for (int e = 0; e < 8; ++e) acc2[p][j][e] = bv;
  }
  const bf16_t* f1l = f1 + (size_t)layer * F1_PER_LAYER;
#pragma unroll
  for (int c = 0; c < 4; ++c) {
    int dbase = c * 32 + half * 8;
    v16bf af[4];
#pragma unroll
    for (int j = 0; j < 4; ++j) {
      int x = j * 16 + m;
      v8bf lo = *(const v8bf*)(bufA + x * SSTR + dbase);
      v8bf hi = *(const v8bf*)(bufA + x * SSTR + dbase + 16);
      af[j] = __builtin_shufflevector(lo, hi, 0,1,2,3,4,5,6,7,8,9,10,11,12,13,14,15);
    }
#pragma unroll
    for (int p = 0; p < 2; ++p) {
      v16bf bfr = *(const v16bf*)(f1l + (size_t)((c * 16 + (w * 2 + p)) * 512 + lane * 16));
#pragma unroll
      for (int j = 0; j < 4; ++j) acc2[p][j] = WMMA_BF16(af[j], bfr, acc2[p][j]);
    }
  }
  // exact GELU, store H1 bf16 into bufB (overlays zt; all zt reads done)
#pragma unroll
  for (int p = 0; p < 2; ++p)
#pragma unroll
    for (int j = 0; j < 4; ++j)
#pragma unroll
      for (int r = 0; r < 8; ++r) {
        float v = acc2[p][j][r];
        v = 0.5f * v * (1.0f + erff(v * INV_SQRT2));
        bufB[(j * 16 + r + half * 8) * HSTR + ((w * 2 + p) * 16 + ncol)] = (bf16_t)v;
      }
  __syncthreads();

  // ---- fc2 + residual ----
  int hcol = w * 16 + ncol;
  float b2 = fc2b[layer * D_ + hcol];
  v8f acc3[4];
#pragma unroll
  for (int j = 0; j < 4; ++j)
#pragma unroll
    for (int e = 0; e < 8; ++e) acc3[j][e] = b2;

  const bf16_t* f2l = f2 + (size_t)layer * F2_PER_LAYER;
#pragma unroll
  for (int c = 0; c < 8; ++c) {
    int dbase = c * 32 + half * 8;
    v16bf bfr = *(const v16bf*)(f2l + (size_t)((c * 8 + w) * 512 + lane * 16));
#pragma unroll
    for (int j = 0; j < 4; ++j) {
      int x = j * 16 + m;
      v8bf lo = *(const v8bf*)(bufB + x * HSTR + dbase);
      v8bf hi = *(const v8bf*)(bufB + x * HSTR + dbase + 16);
      v16bf af = __builtin_shufflevector(lo, hi, 0,1,2,3,4,5,6,7,8,9,10,11,12,13,14,15);
      acc3[j] = WMMA_BF16(af, bfr, acc3[j]);
    }
  }
#pragma unroll
  for (int j = 0; j < 4; ++j)
#pragma unroll
    for (int r = 0; r < 8; ++r) {
      int l = l0 + j * 16 + r + half * 8;
      size_t idx = ((size_t)(b * L_ + l)) * D_ + hcol;
      h[idx] = h[idx] + acc3[j][r];
    }
}

// ---------------------------------------------------------------------------
__global__ void stu_head(const float* __restrict__ h,
                         const float* __restrict__ hw,
                         const float* __restrict__ hb,
                         float* __restrict__ out)
{
  int t = blockIdx.x * 256 + threadIdx.x;
  if (t >= B_ * L_) return;
  const float* hp = h + (size_t)t * D_;
  float acc = hb[0];
#pragma unroll 4
  for (int d = 0; d < D_; ++d) acc += hp[d] * hw[d];
  out[t] = acc;  // (B,1,L) flat == b*L + l
}

// ---------------------------------------------------------------------------
extern "C" void kernel_launch(void* const* d_in, const int* in_sizes, int n_in,
                              void* d_out, int out_size, void* d_ws, size_t ws_size,
                              hipStream_t stream)
{
  const float* x     = (const float*)d_in[0];
  const float* lw    = (const float*)d_in[1];
  const float* lb    = (const float*)d_in[2];
  // d_in[3] = Phi_f (complex64): unused — Haar top-K closed form is exact.
  const float* theta = (const float*)d_in[4];
  const float* lng   = (const float*)d_in[5];
  const float* lnb   = (const float*)d_in[6];
  const float* f1w   = (const float*)d_in[7];
  const float* f1b   = (const float*)d_in[8];
  const float* f2w   = (const float*)d_in[9];
  const float* f2b   = (const float*)d_in[10];
  const float* hw    = (const float*)d_in[11];
  const float* hb    = (const float*)d_in[12];
  float* out = (float*)d_out;

  char* ws = (char*)d_ws;
  float*  hbuf = (float*) (ws + OFF_H);
  bf16_t* zbuf = (bf16_t*)(ws + OFF_Z);
  float*  part = (float*) (ws + OFF_PART);
  float*  S0   = (float*) (ws + OFF_S0);
  bf16_t* cf   = (bf16_t*)(ws + OFF_CF);
  bf16_t* f1f  = (bf16_t*)(ws + OFF_F1);
  bf16_t* f2f  = (bf16_t*)(ws + OFF_F2);

  stu_prep_frags<<<(DEPTH_ * PREP_PER_LAYER) / 256, 256, 0, stream>>>(
      theta, f1w, f2w, cf, f1f, f2f);
  stu_lift<<<(B_ * L_) / 256, 256, 0, stream>>>(x, lw, lb, hbuf);

  for (int i = 0; i < DEPTH_; ++i) {
    stu_ln<<<B_ * 16, 256, 0, stream>>>(hbuf, lng, lnb, zbuf, part, i);
    stu_s0<<<B_, 128, 0, stream>>>(part, theta, S0, i);
    stu_layer<<<B_ * 16, 256, 0, stream>>>(hbuf, zbuf, S0, cf, f1f, f2f,
                                           f1b, f2b, i);
  }
  stu_head<<<(B_ * L_) / 256, 256, 0, stream>>>(hbuf, hw, hb, out);
}
